// CrossLinearAttention_38749194944573
// MI455X (gfx1250) — matrix operand order repaired and checked by reading, not compiled, not run on hardware
//
#include <hip/hip_runtime.h>
#include <hip/hip_bf16.h>

typedef __attribute__((ext_vector_type(16))) __bf16 v16bf;
typedef __attribute__((ext_vector_type(8)))  float  v8f;

// inv_freq[i] = 10000^(-2i/32) = 10^(-i/4)
__constant__ float INV_FREQ[16] = {
    1.0f,    0.5623413252f,  0.3162277660f,  0.1778279410f,
    0.1f,    0.05623413252f, 0.03162277660f, 0.01778279410f,
    0.01f,   0.005623413252f,0.003162277660f,0.001778279410f,
    0.001f,  0.0005623413252f,0.0003162277660f,0.0001778279410f};

// A-matrix (16x32 bf16) K index for packed element j (=2i+p) of lane L.
__device__ __forceinline__ int a_kidx(int j, int lane) {
    int i = j >> 1, p = j & 1;
    return ((i & 4) << 2) + ((lane >> 4) << 3) + ((i & 3) << 1) + p;
}

__device__ __forceinline__ v8f wmma_bf16(v16bf a, v16bf b, v8f c) {
    return __builtin_amdgcn_wmma_f32_16x16x32_bf16(false, a, false, b,
                                                   (short)0, c, false, false);
}

#define BATCH 4
#define N1 8192
#define N2 8192
#define DIM 256
#define HEADS 8
#define INNER 512
#define RB 16          // row-blocks (of 16 rows) processed per block

// ---------------------------------------------------------------------------
// K1: kv = z @ Wkv^T (WMMA, weights held in VGPRs), per-head layernorm,
//     rotary on k. Block: 256 thr / 8 waves, covers a 256-col N-slice
//     (blockIdx.y of 4) and loops over RB row-blocks (blockIdx.x).
// ---------------------------------------------------------------------------
__global__ __launch_bounds__(256) void kv_kernel(
    const float* __restrict__ z, const float* __restrict__ Wkv,
    const float* __restrict__ z_pos,
    const float* __restrict__ k_gamma, const float* __restrict__ k_beta,
    const float* __restrict__ v_gamma, const float* __restrict__ v_beta,
    __bf16* __restrict__ kbf, __bf16* __restrict__ vbf)
{
    __shared__ __align__(32) __bf16 lds_a[16][DIM];   // 8KB
    __shared__ float lds_out[16][256];                // 16KB
    const int tid = threadIdx.x, wave = tid >> 5, lane = tid & 31;
    const int ny = blockIdx.y;                 // 256-col slice (0..3)
    const int nbase = ny * 256 + wave * 32;

    // B panel in registers: 2 N-tiles x 8 k-chunks = 128 VGPRs
    v16bf breg[2][8];
#pragma unroll
    for (int t = 0; t < 2; ++t)
#pragma unroll
        for (int kc = 0; kc < 8; ++kc) {
            v16bf b;
#pragma unroll
            for (int j = 0; j < 16; ++j)
                b[j] = (__bf16)Wkv[(size_t)(nbase + t * 16 + (lane & 15)) * DIM + kc * 32 + a_kidx(j, lane)];
            breg[t][kc] = b;
        }

    for (int rb = 0; rb < RB; ++rb) {
        const int row0 = (blockIdx.x * RB + rb) * 16;
        for (int idx = tid; idx < 16 * DIM; idx += 256) {
            int r = idx >> 8, c = idx & 255;
            lds_a[r][c] = (__bf16)z[(size_t)(row0 + r) * DIM + c];
        }
        if (rb + 1 < RB) {  // prefetch next row-block (16 rows x 1KB)
            const float* pf = z + (size_t)(row0 + 16 + (tid >> 4)) * DIM + (tid & 15) * 16;
            __builtin_prefetch(pf, 0, 1);
        }
        __syncthreads();
#pragma unroll
        for (int t = 0; t < 2; ++t) {
            v8f c = {};
#pragma unroll
            for (int kc = 0; kc < 8; ++kc) {
                v16bf a;
#pragma unroll
                for (int j = 0; j < 16; ++j)
                    a[j] = lds_a[lane & 15][kc * 32 + a_kidx(j, lane)];
                c = wmma_bf16(a, breg[t][kc], c);
            }
#pragma unroll
            for (int i = 0; i < 8; ++i)
                lds_out[i + ((lane >> 4) << 3)][wave * 32 + t * 16 + (lane & 15)] = c[i];
        }
        __syncthreads();
        if (tid < 64) {   // 16 rows x 4 head-groups in this slice
            const int r = tid >> 2, gl = tid & 3;
            const int c0 = gl * 64;
            const int g_abs = ny * 4 + gl;             // 0..15
            const bool is_k = g_abs < 8;
            const int h = g_abs & 7;
            const int grow = row0 + r;
            const int bidx = grow >> 13, n = grow & (N2 - 1);
            const float* gamma = is_k ? k_gamma : v_gamma;
            const float* beta  = is_k ? k_beta  : v_beta;
            float mean = 0.f;
#pragma unroll
            for (int d = 0; d < 64; ++d) mean += lds_out[r][c0 + d];
            mean *= (1.f / 64.f);
            float var = 0.f;
#pragma unroll
            for (int d = 0; d < 64; ++d) {
                float u = lds_out[r][c0 + d] - mean; var += u * u;
            }
            const float rstd = rsqrtf(var * (1.f / 64.f) + 1e-5f);
            __bf16* dst = (is_k ? kbf : vbf) +
                          (((size_t)(bidx * HEADS + h)) * N2 + n) * 64;
            if (is_k) {
                const float px = z_pos[((size_t)bidx * N2 + n) * 2 + 0] * 64.f;
                const float py = z_pos[((size_t)bidx * N2 + n) * 2 + 1] * 64.f;
#pragma unroll
                for (int half = 0; half < 2; ++half) {
                    const float tt = half ? py : px;
#pragma unroll
                    for (int i = 0; i < 16; ++i) {
                        const int d0 = half * 32 + i;
                        const float cf = __cosf(tt * INV_FREQ[i]);
                        const float sf = __sinf(tt * INV_FREQ[i]);
                        const float a0 = (lds_out[r][c0 + d0]      - mean) * rstd * gamma[d0]      + beta[d0];
                        const float a1 = (lds_out[r][c0 + d0 + 16] - mean) * rstd * gamma[d0 + 16] + beta[d0 + 16];
                        dst[d0]      = (__bf16)(a0 * cf - a1 * sf);
                        dst[d0 + 16] = (__bf16)(a1 * cf + a0 * sf);
                    }
                }
            } else {
#pragma unroll
                for (int d = 0; d < 64; ++d)
                    dst[d] = (__bf16)((lds_out[r][c0 + d] - mean) * rstd * gamma[d] + beta[d]);
            }
        }
        __syncthreads();
    }
}

// ---------------------------------------------------------------------------
// K2: q = rotary(x @ Wq^T) -> bf16 [B*N1, 512]. Same scheme, 2 N-slices.
// ---------------------------------------------------------------------------
__global__ __launch_bounds__(256) void q_kernel(
    const float* __restrict__ x, const float* __restrict__ Wq,
    const float* __restrict__ x_pos, __bf16* __restrict__ qbf)
{
    __shared__ __align__(32) __bf16 lds_a[16][DIM];
    __shared__ float lds_out[16][256];
    const int tid = threadIdx.x, wave = tid >> 5, lane = tid & 31;
    const int ny = blockIdx.y;                 // 0..1
    const int nbase = ny * 256 + wave * 32;

    v16bf breg[2][8];
#pragma unroll
    for (int t = 0; t < 2; ++t)
#pragma unroll
        for (int kc = 0; kc < 8; ++kc) {
            v16bf b;
#pragma unroll
            for (int j = 0; j < 16; ++j)
                b[j] = (__bf16)Wq[(size_t)(nbase + t * 16 + (lane & 15)) * DIM + kc * 32 + a_kidx(j, lane)];
            breg[t][kc] = b;
        }

    for (int rb = 0; rb < RB; ++rb) {
        const int row0 = (blockIdx.x * RB + rb) * 16;
        for (int idx = tid; idx < 16 * DIM; idx += 256) {
            int r = idx >> 8, c = idx & 255;
            lds_a[r][c] = (__bf16)x[(size_t)(row0 + r) * DIM + c];
        }
        if (rb + 1 < RB) {
            const float* pf = x + (size_t)(row0 + 16 + (tid >> 4)) * DIM + (tid & 15) * 16;
            __builtin_prefetch(pf, 0, 1);
        }
        __syncthreads();
#pragma unroll
        for (int t = 0; t < 2; ++t) {
            v8f c = {};
#pragma unroll
            for (int kc = 0; kc < 8; ++kc) {
                v16bf a;
#pragma unroll
                for (int j = 0; j < 16; ++j)
                    a[j] = lds_a[lane & 15][kc * 32 + a_kidx(j, lane)];
                c = wmma_bf16(a, breg[t][kc], c);
            }
#pragma unroll
            for (int i = 0; i < 8; ++i)
                lds_out[i + ((lane >> 4) << 3)][wave * 32 + t * 16 + (lane & 15)] = c[i];
        }
        __syncthreads();
        if (tid < 64) {
            const int r = tid >> 2, gl = tid & 3;
            const int c0 = gl * 64;
            const int h = ny * 4 + gl;               // 0..7
            const int grow = row0 + r;
            const int bidx = grow >> 13, n = grow & (N1 - 1);
            const float px = x_pos[((size_t)bidx * N1 + n) * 2 + 0] * 64.f;
            const float py = x_pos[((size_t)bidx * N1 + n) * 2 + 1] * 64.f;
            __bf16* dst = qbf + (size_t)grow * INNER + h * 64;
#pragma unroll
            for (int half = 0; half < 2; ++half) {
                const float tt = half ? py : px;
#pragma unroll
                for (int i = 0; i < 16; ++i) {
                    const int d0 = half * 32 + i;
                    const float cf = __cosf(tt * INV_FREQ[i]);
                    const float sf = __sinf(tt * INV_FREQ[i]);
                    const float a0 = lds_out[r][c0 + d0];
                    const float a1 = lds_out[r][c0 + d0 + 16];
                    dst[d0]      = (__bf16)(a0 * cf - a1 * sf);
                    dst[d0 + 16] = (__bf16)(a1 * cf + a0 * sf);
                }
            }
        }
        __syncthreads();
    }
}

// ---------------------------------------------------------------------------
// K3: dots[bh] (64x64) += K^T V over a 256-row chunk, staged once in LDS.
//     512 thr / 16 waves = 4x4 tile grid. K,V read from HBM exactly once.
// ---------------------------------------------------------------------------
__global__ __launch_bounds__(512) void dots_kernel(
    const __bf16* __restrict__ kbf, const __bf16* __restrict__ vbf,
    float* __restrict__ dots)
{
    __shared__ __align__(32) __bf16 k_lds[256][64];   // 32KB
    __shared__ __align__(32) __bf16 v_lds[256][64];   // 32KB
    const int bh = blockIdx.x, chunk = blockIdx.y;
    const int tid = threadIdx.x, wave = tid >> 5, lane = tid & 31;
    const int d0 = (wave >> 2) * 16, e0 = (wave & 3) * 16;
    const __bf16* kg = kbf + ((size_t)bh * N2 + chunk * 256) * 64;
    const __bf16* vg = vbf + ((size_t)bh * N2 + chunk * 256) * 64;
    {
        const uint4* ks = (const uint4*)kg;
        const uint4* vs = (const uint4*)vg;
        uint4* kd = (uint4*)&k_lds[0][0];
        uint4* vd = (uint4*)&v_lds[0][0];
        for (int i = tid; i < 2048; i += 512) { kd[i] = ks[i]; vd[i] = vs[i]; }
    }
    __syncthreads();
    v8f c = {};
#pragma unroll
    for (int s = 0; s < 8; ++s) {
        const int n0 = s * 32;
        v16bf a;
#pragma unroll
        for (int j = 0; j < 16; ++j)
            a[j] = k_lds[n0 + a_kidx(j, lane)][d0 + (lane & 15)];
        v16bf b;
#pragma unroll
        for (int j = 0; j < 16; ++j)
            b[j] = v_lds[n0 + a_kidx(j, lane)][e0 + (lane & 15)];
        c = wmma_bf16(a, b, c);
    }
    float* dst = dots + ((size_t)bh * 64 + d0) * 64 + e0;
#pragma unroll
    for (int i = 0; i < 8; ++i)
        unsafeAtomicAdd(&dst[(i + ((lane >> 4) << 3)) * 64 + (lane & 15)], c[i]);
}

// ---------------------------------------------------------------------------
// K4: M[b][h*64+d][c] = (1/n2) * sum_e dots[b,h,d,e] * Wo[c, h*64+e]
// ---------------------------------------------------------------------------
__global__ __launch_bounds__(256) void combine_kernel(
    const float* __restrict__ dots, const float* __restrict__ Wo,
    __bf16* __restrict__ Mmat)
{
    const int idx = blockIdx.x * 256 + threadIdx.x;   // 4*512*256
    const int c  = idx & 255;
    const int hd = (idx >> 8) & 511;
    const int bb = idx >> 17;
    const int h = hd >> 6, d = hd & 63;
    const float* drow = dots + (((size_t)bb * HEADS + h) * 64 + d) * 64;
    const float* wrow = Wo + (size_t)c * INNER + h * 64;
    float acc = 0.f;
#pragma unroll
    for (int e = 0; e < 64; ++e) acc += drow[e] * wrow[e];
    Mmat[idx] = (__bf16)(acc * (1.0f / (float)N2));
}

// ---------------------------------------------------------------------------
// K5: out[b] = Q[b] (8192x512) @ M[b] (512x256) + bo. M[b] held in VGPRs
//     (16 waves x 16 cols, 128 VGPRs each); loops RB row-blocks.
// ---------------------------------------------------------------------------
__global__ __launch_bounds__(512) void out_kernel(
    const __bf16* __restrict__ qbf, const __bf16* __restrict__ Mmat,
    const float* __restrict__ bo, float* __restrict__ out)
{
    __shared__ __align__(32) __bf16 lds_a[16][INNER];   // 16KB
    const int tid = threadIdx.x, wave = tid >> 5, lane = tid & 31;
    const int bidx = blockIdx.y;
    const int n0 = wave * 16;
    const __bf16* Mb = Mmat + (size_t)bidx * INNER * DIM;

    v16bf breg[16];
#pragma unroll
    for (int kc = 0; kc < 16; ++kc)
#pragma unroll
        for (int j = 0; j < 16; ++j)
            breg[kc][j] = Mb[(size_t)(kc * 32 + a_kidx(j, lane)) * DIM + n0 + (lane & 15)];

    const float bias = bo[n0 + (lane & 15)];
    for (int rb = 0; rb < RB; ++rb) {
        const int row0 = bidx * N1 + (blockIdx.x * RB + rb) * 16;
        {
            const uint4* src = (const uint4*)(qbf + (size_t)row0 * INNER);
            uint4* dstl = (uint4*)&lds_a[0][0];
            for (int i = tid; i < 1024; i += 512) dstl[i] = src[i];
        }
        if (rb + 1 < RB) {
            const __bf16* pf = qbf + (size_t)(row0 + 16) * INNER + tid * 32;
            __builtin_prefetch(pf, 0, 1);
        }
        __syncthreads();
        v8f c = {};
#pragma unroll
        for (int kc = 0; kc < 16; ++kc) {
            v16bf a;
#pragma unroll
            for (int j = 0; j < 16; ++j)
                a[j] = lds_a[lane & 15][kc * 32 + a_kidx(j, lane)];
            c = wmma_bf16(a, breg[kc], c);
        }
        const int n = n0 + (lane & 15);
#pragma unroll
        for (int i = 0; i < 8; ++i)
            out[(size_t)(row0 + i + ((lane >> 4) << 3)) * DIM + n] = c[i] + bias;
        __syncthreads();
    }
}

extern "C" void kernel_launch(void* const* d_in, const int* in_sizes, int n_in,
                              void* d_out, int out_size, void* d_ws, size_t ws_size,
                              hipStream_t stream) {
    (void)in_sizes; (void)n_in; (void)out_size; (void)ws_size;
    const float* x       = (const float*)d_in[0];
    const float* z       = (const float*)d_in[1];
    const float* x_pos   = (const float*)d_in[2];
    const float* z_pos   = (const float*)d_in[3];
    const float* Wq      = (const float*)d_in[4];
    const float* Wkv     = (const float*)d_in[5];
    const float* k_gamma = (const float*)d_in[6];
    const float* k_beta  = (const float*)d_in[7];
    const float* v_gamma = (const float*)d_in[8];
    const float* v_beta  = (const float*)d_in[9];
    const float* Wo      = (const float*)d_in[10];
    const float* bo      = (const float*)d_in[11];
    float* out = (float*)d_out;

    char* ws = (char*)d_ws;
    __bf16* kbf  = (__bf16*)(ws + 0);           // 32*8192*64 bf16 = 32MB
    __bf16* vbf  = (__bf16*)(ws + 33554432);    // 32MB
    __bf16* qbf  = (__bf16*)(ws + 67108864);    // 32MB
    float*  dots = (float*) (ws + 100663296);   // 32*64*64 f32 = 512KB
    __bf16* Mmat = (__bf16*)(ws + 101187584);   // 4*512*256 bf16 = 1MB

    hipMemsetAsync(dots, 0, (size_t)32 * 64 * 64 * sizeof(float), stream);

    kv_kernel<<<dim3(128, 4), 256, 0, stream>>>(
        z, Wkv, z_pos, k_gamma, k_beta, v_gamma, v_beta, kbf, vbf);

    dots_kernel<<<dim3(32, 32), 512, 0, stream>>>(kbf, vbf, dots);

    combine_kernel<<<(BATCH * INNER * DIM) / 256, 256, 0, stream>>>(dots, Wo, Mmat);

    q_kernel<<<dim3(128, 2), 256, 0, stream>>>(x, Wq, x_pos, qbf);

    out_kernel<<<dim3(32, BATCH), 512, 0, stream>>>(qbf, Mmat, bo, out);
}
